// HeteroGAT_4552665333905
// MI455X (gfx1250) — matrix operand order repaired and checked by reading, not compiled, not run on hardware
//
#include <hip/hip_runtime.h>
#include <hip/hip_bf16.h>
#include <math.h>

typedef __attribute__((ext_vector_type(16))) _Float16 v16h;
typedef __attribute__((ext_vector_type(8)))  _Float16 v8h;
typedef __attribute__((ext_vector_type(8)))  float    v8f;

#define LRELU(x, s) ((x) > 0.0f ? (x) : (s) * (x))

// ---- CDNA5 async global->LDS helpers (ASYNCcnt path, ISA 08_async_tensor §4) ----
__device__ __forceinline__ uint32_t lds_off(const void* p) {
    // generic LDS pointer = {SHARED_BASE[63:32], lds_offset[31:0]}
    return (uint32_t)(uintptr_t)p;
}
__device__ __forceinline__ void async_ld16(uint32_t lds, const _Float16* g) {
    asm volatile("global_load_async_to_lds_b128 %0, %1, off"
                 :: "v"(lds), "v"(g) : "memory");
}
__device__ __forceinline__ void wait_async_le(int n) {
    if (n >= 4)      asm volatile("s_wait_asynccnt 0x4" ::: "memory");
    else if (n >= 2) asm volatile("s_wait_asynccnt 0x2" ::: "memory");
    else             asm volatile("s_wait_asynccnt 0x0" ::: "memory");
}

// =====================================================================
// WMMA GEMM: C[M,128] = A[M,K] @ W[K,128], f16 WMMA / f32 acc.
// Wt is the weight pre-transposed to [128 cols][K] in f16.
// AF16=1: A is f16 [M,K] -> staged via async global->LDS copies.
// AF16=0: A is f32 [M,K] -> staged via VGPR convert (layer-1 features).
// Double-buffered LDS (2 x 8KB per operand), 256 thr = 8 waves,
// 128x128 tile, K step 32; each wave: 16 rows x 128 cols = 8 WMMAs/step.
// =====================================================================
template <int AF16>
__global__ __launch_bounds__(256) void gemm_wmma_n128(
    const void* __restrict__ Ap, const _Float16* __restrict__ Wt,
    float* __restrict__ C, int M, int K)
{
    __shared__ __align__(16) _Float16 As[2][128 * 32];   // As[p][row][k]
    __shared__ __align__(16) _Float16 Bs[2][128 * 32];   // Bs[p][col][k]

    const int tid  = threadIdx.x;
    const int wave = tid >> 5;
    const int lane = tid & 31;
    const int m0   = blockIdx.x * 128;
    const int nt   = K >> 5;

    const int ar   = tid >> 1;         // staged A-row / B-col (0..127)
    const int aseg = (tid & 1) * 16;   // half-offset 0 or 16

    // per-wave uniform: does any lane of this wave stage a valid A row?
    const bool aAny = (m0 + (wave << 4)) < M;

    // stage tile t into buffer p; returns #async instructions issued (per wave)
    auto stage = [&](int t, int p) -> int {
        const int k0 = t << 5;
        int cnt = 2;
        {   // B: Wt[col][k0 .. k0+32) -> Bs[p][col][*]  (64B contiguous per col)
            const _Float16* g = Wt + (size_t)ar * K + k0 + aseg;
            uint32_t l = lds_off(&Bs[p][ar * 32 + aseg]);
            async_ld16(l, g);
            async_ld16(l + 16, g + 8);
        }
        const int row = m0 + ar;
        if (AF16) {
            if (row < M) {
                const _Float16* g = (const _Float16*)Ap + (size_t)row * K + k0 + aseg;
                uint32_t l = lds_off(&As[p][ar * 32 + aseg]);
                async_ld16(l, g);
                async_ld16(l + 16, g + 8);
            } else {
                v8h z = {};
                *(v8h*)&As[p][ar * 32 + aseg]     = z;
                *(v8h*)&As[p][ar * 32 + aseg + 8] = z;
            }
            if (aAny) cnt += 2;
        } else {
            if (row < M) {
                const float* ga = (const float*)Ap + (size_t)row * K + k0 + aseg;
#pragma unroll
                for (int u = 0; u < 16; u += 4) {
                    float4 f = *(const float4*)(ga + u);
                    As[p][ar * 32 + aseg + u + 0] = (_Float16)f.x;
                    As[p][ar * 32 + aseg + u + 1] = (_Float16)f.y;
                    As[p][ar * 32 + aseg + u + 2] = (_Float16)f.z;
                    As[p][ar * 32 + aseg + u + 3] = (_Float16)f.w;
                }
                if (k0 + 32 < K) __builtin_prefetch(ga + 32, 0, 0);
            } else {
                v8h z = {};
                *(v8h*)&As[p][ar * 32 + aseg]     = z;
                *(v8h*)&As[p][ar * 32 + aseg + 8] = z;
            }
        }
        return cnt;
    };

    v8f acc[8];
#pragma unroll
    for (int j = 0; j < 8; ++j) acc[j] = (v8f){};

    stage(0, 0);
    for (int t = 0; t < nt; ++t) {
        const int p = t & 1;
        int pending = 0;
        if (t + 1 < nt) pending = stage(t + 1, p ^ 1);  // prefetch next tile
        wait_async_le(pending);                          // tile t landed in LDS
        __syncthreads();

        // A fragment: lane row = wave*16 + lane%16; K halves {b..b+7,16+b..16+b+7}
        const int mrow  = wave * 16 + (lane & 15);
        const int abase = (lane < 16) ? 0 : 8;
        v8h alo = *(const v8h*)&As[p][mrow * 32 + abase];
        v8h ahi = *(const v8h*)&As[p][mrow * 32 + 16 + abase];
        v16h av;
#pragma unroll
        for (int u = 0; u < 8; ++u) { av[u] = alo[u]; av[u + 8] = ahi[u]; }

        const int kb = (lane < 16) ? 0 : 8;  // B: K halves {b..b+7,16+b..16+b+7}, same as A
#pragma unroll
        for (int j = 0; j < 8; ++j) {
            const int col = j * 16 + (lane & 15);
            v8h blo = *(const v8h*)&Bs[p][col * 32 + kb];
            v8h bhi = *(const v8h*)&Bs[p][col * 32 + 16 + kb];
            v16h bv;
#pragma unroll
            for (int u = 0; u < 8; ++u) { bv[u] = blo[u]; bv[u + 8] = bhi[u]; }
            acc[j] = __builtin_amdgcn_wmma_f32_16x16x32_f16(
                false, av, false, bv, (short)0, acc[j], false, false);
        }
        __syncthreads();  // release buffer p for re-staging
    }

#pragma unroll
    for (int j = 0; j < 8; ++j) {
        const int col = j * 16 + (lane & 15);
#pragma unroll
        for (int r = 0; r < 8; ++r) {
            const int row = m0 + wave * 16 + ((lane < 16) ? r : r + 8);
            if (row < M) C[(size_t)row * 128 + col] = acc[j][r];
        }
    }
}

// ============================ helper kernels ==========================
__device__ __forceinline__ void atomicMaxF(float* addr, float val) {
    int* ai  = (int*)addr;
    int  old = __float_as_int(*addr);
    while (true) {
        float f = __int_as_float(old);
        if (f >= val) break;
        int assumed = old;
        old = atomicCAS(ai, assumed, __float_as_int(val));
        if (old == assumed) break;
    }
}

// weight convert+transpose: W f32 [K][128] -> Wt f16 [128][K]
__global__ void k_cvtT(const float* __restrict__ W, _Float16* __restrict__ Wt, int K) {
    int i = blockIdx.x * blockDim.x + threadIdx.x;
    if (i >= K * 128) return;
    int k = i >> 7, c = i & 127;
    Wt[(size_t)c * K + k] = (_Float16)W[i];
}

// s[row] = dot(h[row, 0:128], a)
__global__ __launch_bounds__(128) void k_rowdot(const float* __restrict__ h,
                                                const float* __restrict__ a,
                                                float* __restrict__ s) {
    __shared__ float red[128];
    const int row = blockIdx.x, tid = threadIdx.x;
    red[tid] = h[(size_t)row * 128 + tid] * a[tid];
    __syncthreads();
#pragma unroll
    for (int w = 64; w > 0; w >>= 1) {
        if (tid < w) red[tid] += red[tid + w];
        __syncthreads();
    }
    if (tid == 0) s[row] = red[0];
}

__global__ void k_fill(float* p, float v, int n) {
    int i = blockIdx.x * blockDim.x + threadIdx.x;
    if (i < n) p[i] = v;
}

__global__ void k_logit_max(const float* __restrict__ ssrc, const float* __restrict__ sdst,
                            const int* __restrict__ src, const int* __restrict__ dst,
                            float* __restrict__ emax, int E) {
    int e = blockIdx.x * blockDim.x + threadIdx.x;
    if (e >= E) return;
    float v = ssrc[src[e]] + sdst[dst[e]];
    v = LRELU(v, 0.2f);
    atomicMaxF(&emax[dst[e]], v);
}

__global__ void k_clean(float* p, int n) {   // empty segments: -inf -> 0
    int i = blockIdx.x * blockDim.x + threadIdx.x;
    if (i < n) { float v = p[i]; if (!(v >= -1.0e37f)) p[i] = 0.0f; }
}

__global__ void k_expsum(const float* __restrict__ ssrc, const float* __restrict__ sdst,
                         const int* __restrict__ src, const int* __restrict__ dst,
                         const float* __restrict__ emax, float* __restrict__ ee,
                         float* __restrict__ den, int E) {
    int e = blockIdx.x * blockDim.x + threadIdx.x;
    if (e >= E) return;
    float v = ssrc[src[e]] + sdst[dst[e]];
    v = LRELU(v, 0.2f);
    float x = __expf(v - emax[dst[e]]);
    ee[e] = x;
    atomicAdd(&den[dst[e]], x);
}

// one wave (32 lanes x float4) per edge: agg[dst] += hs[src] * alpha
__global__ void k_edge_agg(const float* __restrict__ hs,
                           const int* __restrict__ src, const int* __restrict__ dst,
                           const float* __restrict__ ee, const float* __restrict__ den,
                           float* __restrict__ agg, int E) {
    long long gid = (long long)blockIdx.x * blockDim.x + threadIdx.x;
    int e = (int)(gid >> 5);
    if (e >= E) return;
    int lane = (int)(gid & 31);
    int si = src[e], di = dst[e];
    float alpha = ee[e] / (den[di] + 1e-16f);
    const float4 hv = *(const float4*)(hs + (size_t)si * 128 + lane * 4);
    float* ap = agg + (size_t)di * 128 + lane * 4;
    atomicAdd(ap + 0, hv.x * alpha);
    atomicAdd(ap + 1, hv.y * alpha);
    atomicAdd(ap + 2, hv.z * alpha);
    atomicAdd(ap + 3, hv.w * alpha);
}

__global__ void k_addbias(float* __restrict__ p, const float* __restrict__ b, int n) {
    int i = blockIdx.x * blockDim.x + threadIdx.x;
    if (i < n) p[i] += b[i & 127];
}

// leaky-relu f32 -> f16 activations (WMMA consumes f16 next layer)
__global__ void k_leaky_f16(const float* __restrict__ in, _Float16* __restrict__ out,
                            float slope, int n) {
    int i = blockIdx.x * blockDim.x + threadIdx.x;
    if (i < n) { float v = in[i]; out[i] = (_Float16)LRELU(v, slope); }
}

__global__ void k_pool(const _Float16* __restrict__ x, const int* __restrict__ batch,
                       float* __restrict__ pool_t, float* __restrict__ cnt_t, int n) {
    int i = blockIdx.x * blockDim.x + threadIdx.x;
    if (i >= n) return;
    int node = i >> 7, c = i & 127;
    int b = batch[node];
    atomicAdd(&pool_t[(size_t)b * 128 + c], (float)x[i]);
    if (c == 0) atomicAdd(&cnt_t[b], 1.0f);
}

__global__ void k_pool_to_hcat(const float* __restrict__ pool, const float* __restrict__ cnt,
                               _Float16* __restrict__ hcat) {
    int i = blockIdx.x * blockDim.x + threadIdx.x;   // 512 * 640
    if (i >= 512 * 640) return;
    int b = i / 640, col = i % 640;
    int t = col >> 7, c = col & 127;
    float cn = cnt[t * 512 + b];
    hcat[(size_t)b * 2176 + col] =
        (_Float16)(pool[((size_t)t * 512 + b) * 128 + c] / fmaxf(cn, 1.0f));
}

__global__ void k_emb_to_hcat(const float* __restrict__ emb, _Float16* __restrict__ hcat) {
    int i = blockIdx.x * blockDim.x + threadIdx.x;   // 512 * 1536
    if (i >= 512 * 1536) return;
    int b = i / 1536, c = i % 1536;
    hcat[(size_t)b * 2176 + 640 + c] = (_Float16)emb[i];
}

__global__ void k_h1post(float* __restrict__ h1, const float* __restrict__ b1, int n) {
    int i = blockIdx.x * blockDim.x + threadIdx.x;
    if (i < n) { float v = h1[i] + b1[i & 127]; h1[i] = LRELU(v, 0.01f); }
}

__global__ void k_lin2_softmax(const float* __restrict__ h1, const float* __restrict__ w2,
                               const float* __restrict__ b2, float* __restrict__ out) {
    int b = blockIdx.x * blockDim.x + threadIdx.x;
    if (b >= 512) return;
    float o0 = b2[0], o1 = b2[1];
    const float* hr = h1 + (size_t)b * 128;
    for (int c = 0; c < 128; ++c) {
        float h = hr[c];
        o0 += h * w2[c * 2 + 0];
        o1 += h * w2[c * 2 + 1];
    }
    o0 = LRELU(o0, 0.01f);
    o1 = LRELU(o1, 0.01f);
    float m  = fmaxf(o0, o1);
    float e0 = __expf(o0 - m), e1 = __expf(o1 - m);
    float inv = 1.0f / (e0 + e1);
    out[b * 2 + 0] = e0 * inv;
    out[b * 2 + 1] = e1 * inv;
}

// =====================================================================
extern "C" void kernel_launch(void* const* d_in, const int* in_sizes, int n_in,
                              void* d_out, int out_size, void* d_ws, size_t ws_size,
                              hipStream_t stream) {
    (void)n_in; (void)out_size; (void)ws_size;
    static const int SRCT[10] = {3, 3, 3, 4, 4, 0, 1, 2, 0, 1};
    static const int DSTT[10] = {0, 1, 2, 0, 1, 3, 3, 3, 4, 4};

    int N[5];
    for (int t = 0; t < 5; ++t) N[t] = in_sizes[t] / 768;
    int maxN = 0;
    for (int t = 0; t < 5; ++t) if (N[t] > maxN) maxN = N[t];
    int maxE = 0;
    for (int r = 0; r < 10; ++r) { int e = in_sizes[5 + r] / 2; if (e > maxE) maxE = e; }

    // ---- params (jax tree-flatten order: sorted dict keys) ----
    // layers -> lin1_b -> lin1_w -> lin2_b -> lin2_w ; per relation:
    // W_dst, W_src, a_dst, a_src, bias
    const float* P = (const float*)d_in[21];
    const float *Wd[2][10], *Ws[2][10], *Ad[2][10], *Asr[2][10], *Bi[2][10];
    size_t po = 0;
    for (int L = 0; L < 2; ++L) {
        size_t kin = (L == 0) ? 768 : 128;
        for (int r = 0; r < 10; ++r) {
            Wd[L][r]  = P + po; po += kin * 128;
            Ws[L][r]  = P + po; po += kin * 128;
            Ad[L][r]  = P + po; po += 128;
            Asr[L][r] = P + po; po += 128;
            Bi[L][r]  = P + po; po += 128;
        }
    }
    const float* lin1_b = P + po; po += 128;
    const float* lin1_w = P + po; po += (size_t)2176 * 128;
    const float* lin2_b = P + po; po += 2;
    const float* lin2_w = P + po; po += 256;

    // ---- workspace: f32 region then 16B-aligned f16 region ----
    float* w = (float*)d_ws;
    size_t o = 0;
    float* agg[5];  float* aggBase = w + o;
    for (int t = 0; t < 5; ++t) { agg[t] = w + o; o += (size_t)N[t] * 128; }
    size_t aggTotal = (size_t)(w + o - aggBase);
    float* hs   = w + o; o += (size_t)maxN * 128;
    float* hd   = w + o; o += (size_t)maxN * 128;
    float* ssrc = w + o; o += maxN;
    float* sdst = w + o; o += maxN;
    float* emax = w + o; o += maxN;
    float* den  = w + o; o += maxN;
    float* ee   = w + o; o += maxE;
    float* pool = w + o; o += (size_t)5 * 512 * 128;
    float* cnt  = w + o; o += (size_t)5 * 512;
    float* h1   = w + o; o += (size_t)512 * 128;
    o = (o + 3) & ~(size_t)3;                       // 16B align for halves
    _Float16* hb = (_Float16*)(w + o);
    size_t ho = 0;
    _Float16* xh[5];
    for (int t = 0; t < 5; ++t) { xh[t] = hb + ho; ho += (size_t)N[t] * 128; }
    _Float16* hcat = hb + ho; ho += (size_t)512 * 2176;
    _Float16 *WtS[2][10], *WtD[2][10];
    for (int L = 0; L < 2; ++L) {
        size_t kin = (L == 0) ? 768 : 128;
        for (int r = 0; r < 10; ++r) {
            WtS[L][r] = hb + ho; ho += kin * 128;
            WtD[L][r] = hb + ho; ho += kin * 128;
        }
    }
    _Float16* lin1_wt = hb + ho; ho += (size_t)2176 * 128;

    const int T = 256;

    // ---- pre-convert + transpose all weights to f16 Wt[128][K] ----
    for (int L = 0; L < 2; ++L) {
        int kin = (L == 0) ? 768 : 128, n = kin * 128;
        for (int r = 0; r < 10; ++r) {
            k_cvtT<<<(n + T - 1) / T, T, 0, stream>>>(Ws[L][r], WtS[L][r], kin);
            k_cvtT<<<(n + T - 1) / T, T, 0, stream>>>(Wd[L][r], WtD[L][r], kin);
        }
    }
    k_cvtT<<<(2176 * 128 + T - 1) / T, T, 0, stream>>>(lin1_w, lin1_wt, 2176);

    // ======================= GAT layers =======================
    for (int L = 0; L < 2; ++L) {
        const int Kin = (L == 0) ? 768 : 128;
        hipMemsetAsync(aggBase, 0, aggTotal * sizeof(float), stream);
        for (int r = 0; r < 10; ++r) {
            const int s = SRCT[r], d = DSTT[r];
            const int Ns = N[s], Nd = N[d];
            const int E = in_sizes[5 + r] / 2;
            const int* ed   = (const int*)d_in[5 + r];
            const int* esrc = ed;
            const int* edst = ed + E;

            if (L == 0) {
                gemm_wmma_n128<0><<<(Ns + 127) / 128, T, 0, stream>>>(d_in[s], WtS[L][r], hs, Ns, Kin);
                gemm_wmma_n128<0><<<(Nd + 127) / 128, T, 0, stream>>>(d_in[d], WtD[L][r], hd, Nd, Kin);
            } else {
                gemm_wmma_n128<1><<<(Ns + 127) / 128, T, 0, stream>>>(xh[s], WtS[L][r], hs, Ns, Kin);
                gemm_wmma_n128<1><<<(Nd + 127) / 128, T, 0, stream>>>(xh[d], WtD[L][r], hd, Nd, Kin);
            }
            k_rowdot<<<Ns, 128, 0, stream>>>(hs, Asr[L][r], ssrc);
            k_rowdot<<<Nd, 128, 0, stream>>>(hd, Ad[L][r], sdst);

            k_fill<<<(Nd + T - 1) / T, T, 0, stream>>>(emax, -INFINITY, Nd);
            hipMemsetAsync(den, 0, (size_t)Nd * sizeof(float), stream);
            k_logit_max<<<(E + T - 1) / T, T, 0, stream>>>(ssrc, sdst, esrc, edst, emax, E);
            k_clean<<<(Nd + T - 1) / T, T, 0, stream>>>(emax, Nd);
            k_expsum<<<(E + T - 1) / T, T, 0, stream>>>(ssrc, sdst, esrc, edst, emax, ee, den, E);
            long long nth = (long long)E * 32;
            k_edge_agg<<<(unsigned)((nth + T - 1) / T), T, 0, stream>>>(hs, esrc, edst, ee, den, agg[d], E);
            k_addbias<<<((size_t)Nd * 128 + T - 1) / T, T, 0, stream>>>(agg[d], Bi[L][r], Nd * 128);
        }
        for (int t = 0; t < 5; ++t) {
            int n = N[t] * 128;
            k_leaky_f16<<<(n + T - 1) / T, T, 0, stream>>>(agg[t], xh[t], 0.01f, n);
        }
    }

    // ======================= pooling + head =======================
    hipMemsetAsync(pool, 0, ((size_t)5 * 512 * 128 + 5 * 512) * sizeof(float), stream);
    for (int t = 0; t < 5; ++t) {
        int n = N[t] * 128;
        k_pool<<<(n + T - 1) / T, T, 0, stream>>>(
            xh[t], (const int*)d_in[15 + t],
            pool + (size_t)t * 512 * 128, cnt + (size_t)t * 512, n);
    }
    k_pool_to_hcat<<<(512 * 640 + T - 1) / T, T, 0, stream>>>(pool, cnt, hcat);
    k_emb_to_hcat<<<(512 * 1536 + T - 1) / T, T, 0, stream>>>((const float*)d_in[20], hcat);

    gemm_wmma_n128<1><<<(512 + 127) / 128, T, 0, stream>>>(hcat, lin1_wt, h1, 512, 2176);
    k_h1post<<<(512 * 128 + T - 1) / T, T, 0, stream>>>(h1, lin1_b, 512 * 128);
    k_lin2_softmax<<<2, 256, 0, stream>>>(h1, lin2_w, lin2_b, (float*)d_out);
}
